// SparseAttnLayer_49357764166120
// MI455X (gfx1250) — matrix operand order repaired and checked by reading, not compiled, not run on hardware
//
#include <hip/hip_runtime.h>
#include <hip/hip_bf16.h>

// ---------------------------------------------------------------------------
// SparseAttnLayer for MI455X (gfx1250, wave32, WMMA, bf16 data paths)
// B=8, T=2048, D=256, H=4, hd=64
// ---------------------------------------------------------------------------

#define BB   8
#define TT   2048
#define DD   256
#define HH   4
#define HDIM 64
#define NEGV (-1e9f)
#define EPSV 1e-5f
#define INV_SCALE 0.125f   // 1/sqrt(64)

typedef __attribute__((ext_vector_type(16))) __bf16 v16bf;
typedef __attribute__((ext_vector_type(8)))  __bf16 v8bf;
typedef __attribute__((ext_vector_type(8)))  float  v8f;

__device__ __forceinline__ v8f wmma_bf16(v16bf a, v16bf b, v8f c) {
    // D = A(16x32 bf16) * B(32x16 bf16) + C(16x16 f32)
    return __builtin_amdgcn_wmma_f32_16x16x32_bf16(
        /*neg_a=*/false, a, /*neg_b=*/false, b,
        /*c_mod=*/(short)0, c, /*reuse_a=*/false, /*reuse_b=*/false);
}

// A-fragment (16x32 bf16) from a row-major bf16 matrix; row m holds K contiguous.
// wave32 layout: lanes 0-15 -> M=lane, K = k0..k0+15
//                lanes 16-31 -> M=lane-16, K = k0+16..k0+31
__device__ __forceinline__ v16bf a_frag_bf(const __bf16* base, int ld,
                                           int m0, int k0, int lane) {
    const __bf16* p = base + (size_t)(m0 + (lane & 15)) * ld + k0 + ((lane >> 4) << 4);
    const v8bf lo = *(const v8bf*)p;          // 16B load
    const v8bf hi = *(const v8bf*)(p + 8);    // 16B load
    v16bf a;
#pragma unroll
    for (int e = 0; e < 8; ++e) { a[e] = lo[e]; a[e + 8] = hi[e]; }
    return a;
}

// B-fragment (32x16 bf16); SOURCE row n holds the K-dim contiguously
// (X*W^T GEMMs: B[k][n] = W[n][k];  A*V with transposed V: B[k][n] = Vt[n][k]).
// wave32 layout: lane&15 -> N; lanes 0-15 hold K=k0..k0+15, lanes 16-31 K=k0+16..k0+31
__device__ __forceinline__ v16bf b_frag_bf(const __bf16* base, int ld,
                                           int n0, int k0, int lane) {
    const __bf16* p = base + (size_t)(n0 + (lane & 15)) * ld + k0 + ((lane >> 4) << 4);
    return *(const v16bf*)p;                  // 32B -> two b128 loads
}

__device__ __forceinline__ float wred_sum(float v) {
#pragma unroll
    for (int off = 16; off > 0; off >>= 1) v += __shfl_xor(v, off, 32);
    return v;
}
__device__ __forceinline__ float wred_max(float v) {
#pragma unroll
    for (int off = 16; off > 0; off >>= 1) v = fmaxf(v, __shfl_xor(v, off, 32));
    return v;
}

// ---------------------------------------------------------------------------
// Kernel 0: one-time f32 -> bf16 conversion (x and the four weight matrices).
// ---------------------------------------------------------------------------
__global__ void f32_to_bf16_kernel(const float* __restrict__ src,
                                   __bf16* __restrict__ dst, int n) {
    const int i = (blockIdx.x * blockDim.x + threadIdx.x) * 4;
    if (i >= n) return;
    const float4 f = *(const float4*)(src + i);
    dst[i + 0] = (__bf16)f.x;
    dst[i + 1] = (__bf16)f.y;
    dst[i + 2] = (__bf16)f.z;
    dst[i + 3] = (__bf16)f.w;
}

// ---------------------------------------------------------------------------
// Kernel 1: fused QKV projection.  Q/K/V = x @ W{q,k,v}^T  (16384x256 * 256x256)
// One wave per 16x16 output tile; shared A-fragment across the three GEMMs.
// Q,K stored bf16 row-major; V stored bf16 TRANSPOSED per head: Vt[b,h][d][t].
// ---------------------------------------------------------------------------
__global__ void qkv_kernel(const __bf16* __restrict__ x,
                           const __bf16* __restrict__ Wq,
                           const __bf16* __restrict__ Wk,
                           const __bf16* __restrict__ Wv,
                           __bf16* __restrict__ Q,
                           __bf16* __restrict__ K,
                           __bf16* __restrict__ Vt) {
    const int lane = threadIdx.x & 31;
    const int wave = threadIdx.x >> 5;
    const int task = blockIdx.x * 8 + wave;         // (BB*TT/16)*(DD/16) tasks
    const int m0 = (task >> 4) * 16;
    const int n0 = (task & 15) * 16;

    v8f cq = {}, ck = {}, cv = {};
#pragma unroll
    for (int k0 = 0; k0 < DD; k0 += 32) {
        v16bf a  = a_frag_bf(x, DD, m0, k0, lane);
        v16bf bq = b_frag_bf(Wq, DD, n0, k0, lane);
        v16bf bk = b_frag_bf(Wk, DD, n0, k0, lane);
        v16bf bv = b_frag_bf(Wv, DD, n0, k0, lane);
        cq = wmma_bf16(a, bq, cq);
        ck = wmma_bf16(a, bk, ck);
        cv = wmma_bf16(a, bv, cv);
    }

    const int n  = n0 + (lane & 15);    // output column 0..255
    const int hi = lane >> 4;
    const int h  = n >> 6;              // head
    const int d  = n & (HDIM - 1);      // dim within head
#pragma unroll
    for (int r = 0; r < 8; ++r) {
        const int row = m0 + r + 8 * hi;           // global token row = b*T + t
        const int b   = row >> 11;
        const int t   = row & (TT - 1);
        const size_t idx = (size_t)row * DD + n;
        Q[idx] = (__bf16)cq[r];
        K[idx] = (__bf16)ck[r];
        Vt[((size_t)(b * HH + h) * HDIM + d) * TT + t] = (__bf16)cv[r];
    }
}

// ---------------------------------------------------------------------------
// Kernel 2: attention with exact sparsemax (threshold bisection).
// One workgroup (8 waves) per (b, h, 16-query tile).
//   Phase 1: S[16][2048] (f32) = mask(Q K^T / 8)            -> LDS (128 KB)
//   Phase 2: sparsemax per row via tau bisection (register resident),
//            result written to Sbf[16][2048] (bf16)         -> LDS (64 KB)
//   Phase 3: out[16][64] = Sbf @ V  (B-fragments from transposed V)
// ---------------------------------------------------------------------------
__global__ void attn_kernel(const __bf16* __restrict__ Q,
                            const __bf16* __restrict__ K,
                            const __bf16* __restrict__ Vt,
                            const unsigned char* __restrict__ kpm,
                            __bf16* __restrict__ AO) {
    extern __shared__ float smem[];
    float*  S    = smem;                        // 16*TT f32
    __bf16* Sbf  = (__bf16*)(smem + 16 * TT);   // 16*TT bf16
    float*  part = (float*)(Sbf + 16 * TT);     // 8*16*16 f32 partials

    const int lane = threadIdx.x & 31;
    const int wave = threadIdx.x >> 5;
    const int blk  = blockIdx.x;                // BB*HH*(TT/16)
    const int qt   = blk & (TT / 16 - 1);
    const int h    = (blk >> 7) & (HH - 1);
    const int b    = blk >> 9;
    const int t0   = qt * 16;

    const __bf16* Qb  = Q  + (size_t)b * TT * DD + (size_t)h * HDIM;
    const __bf16* Kb  = K  + (size_t)b * TT * DD + (size_t)h * HDIM;
    const __bf16* Vtb = Vt + (size_t)(b * HH + h) * HDIM * TT;
    const unsigned char* mrow = kpm + (size_t)b * TT;

    const int n  = lane & 15;
    const int hi = lane >> 4;

    // ---- Phase 1: scores. Q fragments loaded once, reused over 16 key tiles.
    const v16bf qa0 = a_frag_bf(Qb, DD, t0, 0,  lane);
    const v16bf qa1 = a_frag_bf(Qb, DD, t0, 32, lane);

    for (int kt = 0; kt < 16; ++kt) {
        const int key0 = wave * 256 + kt * 16;
        v16bf kb0 = b_frag_bf(Kb, DD, key0, 0,  lane);   // K rows hold hd contiguous
        v16bf kb1 = b_frag_bf(Kb, DD, key0, 32, lane);
        v8f c = {};
        c = wmma_bf16(qa0, kb0, c);
        c = wmma_bf16(qa1, kb1, c);

        const int key = key0 + n;
        const bool pad = (mrow[key] != 0);
#pragma unroll
        for (int r = 0; r < 8; ++r) {
            float v = c[r] * INV_SCALE;
            if (pad) v = NEGV;
            S[(r + 8 * hi) * TT + key] = v;
        }
    }
    __syncthreads();

    // ---- Phase 2: sparsemax per row. 2 rows per wave, 64 values per lane.
#pragma unroll 1
    for (int rr = 0; rr < 2; ++rr) {
        const int row = wave * 2 + rr;
        const float* srow = S + (size_t)row * TT;
        float v[64];
#pragma unroll
        for (int j = 0; j < 64; ++j) v[j] = srow[j * 32 + lane];

        float mx = v[0];
#pragma unroll
        for (int j = 1; j < 64; ++j) mx = fmaxf(mx, v[j]);
        mx = wred_max(mx);

        // tau in [mx-1, mx]; f(tau)=sum(max(z-tau,0)) monotone decreasing, f(tau*)=1
        float lo = mx - 1.0f, hi2 = mx;
#pragma unroll 1
        for (int it = 0; it < 30; ++it) {
            const float tau = 0.5f * (lo + hi2);
            float s = 0.0f;
#pragma unroll
            for (int j = 0; j < 64; ++j) s += fmaxf(v[j] - tau, 0.0f);
            s = wred_sum(s);
            if (s > 1.0f) lo = tau; else hi2 = tau;
        }
        const float tau = 0.5f * (lo + hi2);
        __bf16* drow = Sbf + (size_t)row * TT;
#pragma unroll
        for (int j = 0; j < 64; ++j)
            drow[j * 32 + lane] = (__bf16)fmaxf(v[j] - tau, 0.0f);
    }
    __syncthreads();

    // ---- Phase 3: out = A @ V.  wave -> (hd tile = wave&3, key half = wave>>2)
    const int ntile = wave & 3;
    const int khalf = wave >> 2;
    v8f c = {};
    for (int kt = 0; kt < 32; ++kt) {
        const int k0 = khalf * 1024 + kt * 32;
        v16bf a  = a_frag_bf(Sbf, TT, 0, k0, lane);            // bf16 from LDS
        v16bf vb = b_frag_bf(Vtb, TT, ntile * 16, k0, lane);   // contiguous (V^T)
        c = wmma_bf16(a, vb, c);
    }
#pragma unroll
    for (int r = 0; r < 8; ++r)
        part[wave * 256 + (r + 8 * hi) * 16 + n] = c[r];
    __syncthreads();

    if (wave < 4) {
#pragma unroll
        for (int r = 0; r < 8; ++r) {
            const int m = r + 8 * hi;
            const float vsum = part[wave * 256 + m * 16 + n] +
                               part[(wave + 4) * 256 + m * 16 + n];
            AO[((size_t)b * TT + t0 + m) * DD + h * HDIM + wave * 16 + n] =
                (__bf16)vsum;
        }
    }
}

// ---------------------------------------------------------------------------
// Kernel 3: y = attn_out @ Wo^T + x, then LayerNorm(gamma, beta).  (f32 out)
// ---------------------------------------------------------------------------
__global__ void proj_ln_kernel(const __bf16* __restrict__ AO,
                               const float* __restrict__ x,
                               const __bf16* __restrict__ Wo,
                               const float* __restrict__ gamma,
                               const float* __restrict__ beta,
                               float* __restrict__ out) {
    __shared__ float y[16 * DD];
    const int lane = threadIdx.x & 31;
    const int wave = threadIdx.x >> 5;
    const int m0 = blockIdx.x * 16;
    const int n0a = wave * 32;
    const int n0b = wave * 32 + 16;

    v8f c0 = {}, c1 = {};
#pragma unroll
    for (int k0 = 0; k0 < DD; k0 += 32) {
        v16bf a  = a_frag_bf(AO, DD, m0, k0, lane);
        v16bf b0 = b_frag_bf(Wo, DD, n0a, k0, lane);
        v16bf b1 = b_frag_bf(Wo, DD, n0b, k0, lane);
        c0 = wmma_bf16(a, b0, c0);
        c1 = wmma_bf16(a, b1, c1);
    }

    const int n  = lane & 15;
    const int hi = lane >> 4;
#pragma unroll
    for (int r = 0; r < 8; ++r) {
        const int m = r + 8 * hi;
        y[m * DD + n0a + n] = c0[r] + x[(size_t)(m0 + m) * DD + n0a + n];
        y[m * DD + n0b + n] = c1[r] + x[(size_t)(m0 + m) * DD + n0b + n];
    }
    __syncthreads();

#pragma unroll 1
    for (int rr = 0; rr < 2; ++rr) {
        const int row = wave * 2 + rr;
        float v[8];
#pragma unroll
        for (int j = 0; j < 8; ++j) v[j] = y[row * DD + j * 32 + lane];

        float s = 0.0f;
#pragma unroll
        for (int j = 0; j < 8; ++j) s += v[j];
        const float mu = wred_sum(s) * (1.0f / DD);

        float q = 0.0f;
#pragma unroll
        for (int j = 0; j < 8; ++j) { const float d = v[j] - mu; q += d * d; }
        const float var = wred_sum(q) * (1.0f / DD);
        const float inv = rsqrtf(var + EPSV);

#pragma unroll
        for (int j = 0; j < 8; ++j) {
            const int col = j * 32 + lane;
            out[(size_t)(m0 + row) * DD + col] =
                (v[j] - mu) * inv * gamma[col] + beta[col];
        }
    }
}

// ---------------------------------------------------------------------------
extern "C" void kernel_launch(void* const* d_in, const int* in_sizes, int n_in,
                              void* d_out, int out_size, void* d_ws, size_t ws_size,
                              hipStream_t stream) {
    const float*         x     = (const float*)d_in[0];
    const unsigned char* kpm   = (const unsigned char*)d_in[1];  // jax bool: 1 byte
    const float*         Wq    = (const float*)d_in[2];
    const float*         Wk    = (const float*)d_in[3];
    const float*         Wv    = (const float*)d_in[4];
    const float*         Wo    = (const float*)d_in[5];
    const float*         gamma = (const float*)d_in[6];
    const float*         beta  = (const float*)d_in[7];
    float*               out   = (float*)d_out;

    const size_t NTD = (size_t)BB * TT * DD;   // 16384*256 = 4,194,304
    const size_t WSZ = (size_t)DD * DD;        // 65,536

    __bf16* wsb  = (__bf16*)d_ws;
    __bf16* xbf  = wsb;                         // NTD
    __bf16* Wqb  = xbf + NTD;                   // WSZ
    __bf16* Wkb  = Wqb + WSZ;
    __bf16* Wvb  = Wkb + WSZ;
    __bf16* Wob  = Wvb + WSZ;
    __bf16* Qbf  = Wob + WSZ;                   // NTD
    __bf16* Kbf  = Qbf + NTD;                   // NTD
    __bf16* Vtb  = Kbf + NTD;                   // NTD (transposed per head)
    __bf16* AObf = Vtb + NTD;                   // NTD

    // One-time f32 -> bf16 conversions.
    f32_to_bf16_kernel<<<dim3((NTD / 4 + 255) / 256), dim3(256), 0, stream>>>(x,  xbf, (int)NTD);
    f32_to_bf16_kernel<<<dim3((WSZ / 4 + 255) / 256), dim3(256), 0, stream>>>(Wq, Wqb, (int)WSZ);
    f32_to_bf16_kernel<<<dim3((WSZ / 4 + 255) / 256), dim3(256), 0, stream>>>(Wk, Wkb, (int)WSZ);
    f32_to_bf16_kernel<<<dim3((WSZ / 4 + 255) / 256), dim3(256), 0, stream>>>(Wv, Wvb, (int)WSZ);
    f32_to_bf16_kernel<<<dim3((WSZ / 4 + 255) / 256), dim3(256), 0, stream>>>(Wo, Wob, (int)WSZ);

    // QKV projection: (BB*TT/16)*(DD/16) = 16384 wave-tiles, 8 waves/block.
    qkv_kernel<<<dim3(2048), dim3(256), 0, stream>>>(xbf, Wqb, Wkb, Wvb,
                                                     Qbf, Kbf, Vtb);

    // Attention: BB*HH*(TT/16) = 4096 workgroups,
    // dynamic LDS: 128KB (S f32) + 64KB (Sbf) + 8KB (partials) = 200KB.
    const size_t smem_bytes = (size_t)16 * TT * 4 + (size_t)16 * TT * 2
                            + (size_t)8 * 16 * 16 * 4;
    attn_kernel<<<dim3(BB * HH * (TT / 16)), dim3(256), smem_bytes, stream>>>(
        Qbf, Kbf, Vtb, kpm, AObf);

    // Output projection + residual + LayerNorm: (BB*TT)/16 = 1024 workgroups.
    proj_ln_kernel<<<dim3((BB * TT) / 16), dim3(256), 0, stream>>>(
        AObf, x, Wob, gamma, beta, out);
}
